// RelLearnableMultiHeadAttn_8083128451278
// MI455X (gfx1250) — matrix operand order repaired and checked by reading, not compiled, not run on hardware
//
#include <hip/hip_runtime.h>
#include <hip/hip_bf16.h>

typedef unsigned short u16;
typedef __attribute__((ext_vector_type(16))) __bf16 v16bf;
typedef __attribute__((ext_vector_type(8)))  float   v8f;
typedef __attribute__((ext_vector_type(16))) unsigned short v16u;
typedef __attribute__((ext_vector_type(8)))  unsigned short v8u;

union FragU { v16u v; v8u h[2]; u16 u[16]; };

struct ChunkFrags {           // all fragments for one 32-wide j-chunk of attention
    v16u kB[2][2];            // [sub-tile t][dh-chunk]
    v16u eB[2][2];            // r_emb (shifted) fragments
    v16u vB[4];               // V^T fragments for P@V
    float rb[2];              // r_bias[j+1] per sub-tile column
};

static __device__ __forceinline__ u16 f2bf(float f){
    unsigned u = __builtin_bit_cast(unsigned, f);
    unsigned r = u + 0x7FFFu + ((u >> 16) & 1u);
    return (u16)(r >> 16);
}

static __device__ __forceinline__ v8f vzero8(){
    v8f z;
#pragma unroll
    for (int i = 0; i < 8; ++i) z[i] = 0.f;
    return z;
}

static __device__ __forceinline__ v8f wmma_bf16(v16u a, v16u b, v8f c){
    return __builtin_amdgcn_wmma_f32_16x16x32_bf16(
        false, __builtin_bit_cast(v16bf, a),
        false, __builtin_bit_cast(v16bf, b),
        (short)0, c, false, false);
}

// A fragment (16x32, bf16): lane m = L%16, kb = (L/16)*8; 16 contiguous K starting at 2*kb.
static __device__ __forceinline__ v16u loadA32(const u16* row, int k0, int kb){
    FragU f;
    f.h[0] = *(const v8u*)(row + k0 + 2 * kb);
    f.h[1] = *(const v8u*)(row + k0 + 2 * kb + 8);
    return f.v;
}

// B fragment (32x16, bf16): lane n = L%16, 16 contiguous K starting at (L/16)*16.
static __device__ __forceinline__ v16u loadB32(const u16* p){
    FragU f;
    f.h[0] = *(const v8u*)(p);
    f.h[1] = *(const v8u*)(p + 8);
    return f.v;
}

static __device__ __forceinline__ float rmax16(float x){
    x = fmaxf(x, __shfl_xor(x, 1));
    x = fmaxf(x, __shfl_xor(x, 2));
    x = fmaxf(x, __shfl_xor(x, 4));
    x = fmaxf(x, __shfl_xor(x, 8));
    return x;
}
static __device__ __forceinline__ float rsum16(float x){
    x += __shfl_xor(x, 1);
    x += __shfl_xor(x, 2);
    x += __shfl_xor(x, 4);
    x += __shfl_xor(x, 8);
    return x;
}

// ---------------------------------------------------------------- K0: converts
__global__ void __launch_bounds__(256) k_cvt_bf16(const float* __restrict__ src,
                                                  u16* __restrict__ dst, int n){
    int i = blockIdx.x * 256 + threadIdx.x;
    if (i < n) dst[i] = f2bf(src[i]);
}

// src [rows][cols] f32 -> dst [cols][rows] bf16
__global__ void __launch_bounds__(256) k_transpose_cvt(const float* __restrict__ src,
                                                       u16* __restrict__ dst,
                                                       int rows, int cols){
    int c = blockIdx.x * 16 + (threadIdx.x & 15);
    int r = blockIdx.y * 16 + (threadIdx.x >> 4);
    if (r < rows && c < cols) dst[(size_t)c * rows + r] = f2bf(src[(size_t)r * cols + c]);
}

// ------------------------------------------------- K1: fused QKV projection GEMM
__global__ void __launch_bounds__(256, 1) gemm_qkv(
    const u16* __restrict__ wbf,
    const u16* __restrict__ WqT, const u16* __restrict__ WkT, const u16* __restrict__ WvT,
    const float* __restrict__ rwb,
    u16* __restrict__ qbf, u16* __restrict__ rwbf, u16* __restrict__ kbf, u16* __restrict__ vTbf)
{
    const int tid  = threadIdx.x;
    const int wv   = tid >> 5, lane = tid & 31;
    const int l16  = lane & 15, hl = lane >> 4;
    const int rb0  = blockIdx.x * 64 + (wv & 3) * 16;   // 16-row strip
    const int cb   = blockIdx.y;                        // 0..11 over virtual 3072 cols
    const int sel  = cb >> 2;                           // 0=q 1=k 2=v
    const int cm0  = (cb & 3) * 256 + (wv >> 2) * 128;  // col base within matrix
    const u16* WT  = (sel == 0) ? WqT : (sel == 1) ? WkT : WvT;

    const u16* arow = wbf + (size_t)(rb0 + l16) * 1024;
    const int kb = hl * 8;

    auto loadB8 = [&](v16u* B, int kk){
#pragma unroll
        for (int t = 0; t < 8; ++t){
            const u16* bp = WT + (size_t)(cm0 + t * 16 + l16) * 1024 + kk + hl * 16;
            B[t] = loadB32(bp);
        }
    };

    v8f acc[8];
#pragma unroll
    for (int t = 0; t < 8; ++t) acc[t] = vzero8();

    // software-pipelined: fragments consumed one stage after they are loaded
    v16u A0 = loadA32(arow, 0, kb), A1;
    v16u B0[8], B1[8];
    loadB8(B0, 0);
    for (int kk = 0; kk < 1024; kk += 64){
        A1 = loadA32(arow, kk + 32, kb);
        loadB8(B1, kk + 32);
#pragma unroll
        for (int t = 0; t < 8; ++t) acc[t] = wmma_bf16(A0, B0[t], acc[t]);
        if (kk + 64 < 1024){
            A0 = loadA32(arow, kk + 64, kb);
            loadB8(B0, kk + 64);
        }
#pragma unroll
        for (int t = 0; t < 8; ++t) acc[t] = wmma_bf16(A1, B1[t], acc[t]);
    }

#pragma unroll
    for (int t = 0; t < 8; ++t){
        int col = cm0 + t * 16 + l16;       // 0..1023 within matrix = n*64+dh
        int n = col >> 6, dh = col & 63;
#pragma unroll
        for (int v = 0; v < 8; ++v){
            int g = rb0 + v + 8 * hl;       // global token row
            int b = g >> 10, i = g & 1023;
            float f = acc[t][v];
            size_t hidx = ((size_t)(b * 16 + n) * 1024 + i) * 64 + dh;
            if (sel == 0){
                qbf[hidx]  = f2bf(f);
                rwbf[hidx] = f2bf(f + rwb[col]);          // q + r_w_bias
            } else if (sel == 1){
                kbf[hidx] = f2bf(f);
            } else {
                // V stored transposed: [b,n,dh,j] so P@V B-fragment is contiguous in j
                vTbf[((size_t)(b * 16 + n) * 64 + dh) * 1024 + i] = f2bf(f);
            }
        }
    }
}

// --------------------------------------------- K2: fused flash attention per head
// score[i,j] = ( (q+rwb)·k[j] + q·r_emb[j+1] + r_bias[j+1,n] ) * 0.125 ; last col BD=0
__global__ void __launch_bounds__(256, 1) attn_flash(
    const u16* __restrict__ qbf, const u16* __restrict__ rwbf,
    const u16* __restrict__ kbf, const u16* __restrict__ vTbf,
    const u16* __restrict__ rembbf, const float* __restrict__ rbias,
    u16* __restrict__ attnbf)
{
    __shared__ float ldsP[8][16][33];   // per-wave 16x32 P staging (+1 pad)

    const int tid = threadIdx.x;
    const int wv  = tid >> 5, lane = tid & 31;
    const int l16 = lane & 15, hl = lane >> 4;

    const int blk = blockIdx.x;
    const int it = blk & 7, n = (blk >> 3) & 15, b = blk >> 7;

    const int i0 = it * 128 + wv * 16;
    const size_t head = (size_t)(b * 16 + n) * 1024;
    const int kb = hl * 8;

    const u16* qrow = qbf  + (head + i0 + l16) * 64;
    const u16* rrow = rwbf + (head + i0 + l16) * 64;
    v16u qA0 = loadA32(qrow, 0, kb);
    v16u qA1 = loadA32(qrow, 32, kb);
    v16u rA0 = loadA32(rrow, 0, kb);
    v16u rA1 = loadA32(rrow, 32, kb);

    const u16* kbase = kbf + head * 64;
    const u16* vbase = vTbf + (size_t)(b * 16 + n) * 64 * 1024;

    float m[8], l[8];
    v8f O[4];
#pragma unroll
    for (int c = 0; c < 4; ++c) O[c] = vzero8();
#pragma unroll
    for (int v = 0; v < 8; ++v){ m[v] = -1e30f; l[v] = 0.f; }

    auto loadChunk = [&](int j0) -> ChunkFrags {
        ChunkFrags F;
#pragma unroll
        for (int t = 0; t < 2; ++t){
            const int jc = j0 + t * 16 + l16;
            const u16* kr = kbase + (size_t)jc * 64 + hl * 16;
            F.kB[t][0] = loadB32(kr);
            F.kB[t][1] = loadB32(kr + 32);
            const int jr = jc + 1;                 // rel_shift: column j+1
            if (jr < 1024){
                const u16* er = rembbf + (size_t)jr * 1024 + n * 64 + hl * 16;
                F.eB[t][0] = loadB32(er);
                F.eB[t][1] = loadB32(er + 32);
                F.rb[t] = rbias[jr * 16 + n];
            } else {
                FragU z;
#pragma unroll
                for (int e = 0; e < 16; ++e) z.u[e] = 0;
                F.eB[t][0] = z.v; F.eB[t][1] = z.v; F.rb[t] = 0.f;
            }
        }
#pragma unroll
        for (int c = 0; c < 4; ++c)
            F.vB[c] = loadB32(vbase + (size_t)(c * 16 + l16) * 1024 + j0 + hl * 16);
        return F;
    };

    auto compute = [&](const ChunkFrags& F){
        v8f s0, s1;
#pragma unroll
        for (int t = 0; t < 2; ++t){
            v8f acc = vzero8();
            acc = wmma_bf16(rA0, F.kB[t][0], acc);
            acc = wmma_bf16(rA1, F.kB[t][1], acc);
            acc = wmma_bf16(qA0, F.eB[t][0], acc);
            acc = wmma_bf16(qA1, F.eB[t][1], acc);
            v8f sc;
#pragma unroll
            for (int v = 0; v < 8; ++v) sc[v] = (acc[v] + F.rb[t]) * 0.125f;
            if (t == 0) s0 = sc; else s1 = sc;
        }

        // online softmax (rows live in VGPR index, cols across 16-lane halves)
#pragma unroll
        for (int v = 0; v < 8; ++v){
            float c = fmaxf(s0[v], s1[v]);
            c = rmax16(c);
            float mn = fmaxf(m[v], c);
            float al = __expf(m[v] - mn);
            m[v] = mn;
            float p0 = __expf(s0[v] - mn);
            float p1 = __expf(s1[v] - mn);
            s0[v] = p0; s1[v] = p1;
            float rs = rsum16(p0 + p1);
            l[v] = l[v] * al + rs;
            O[0][v] *= al; O[1][v] *= al; O[2][v] *= al; O[3][v] *= al;
        }

        // transpose P (C-layout -> A-layout) through per-wave LDS
#pragma unroll
        for (int v = 0; v < 8; ++v){
            ldsP[wv][v + 8 * hl][l16]      = s0[v];
            ldsP[wv][v + 8 * hl][16 + l16] = s1[v];
        }
        asm volatile("s_wait_dscnt 0" ::: "memory");   // in-wave cross-lane RAW
        FragU pA;
#pragma unroll
        for (int e = 0; e < 8; ++e){
            pA.u[e]     = f2bf(ldsP[wv][l16][2 * kb + e]);
            pA.u[e + 8] = f2bf(ldsP[wv][l16][2 * kb + 8 + e]);
        }
        asm volatile("" ::: "memory");

        // O += P @ V  (V^T layout -> contiguous in j; fragments preloaded)
#pragma unroll
        for (int c = 0; c < 4; ++c)
            O[c] = wmma_bf16(pA.v, F.vB[c], O[c]);
    };

    // ping-pong pipelined over 64-wide super-chunks: loads for the next chunk
    // are in flight while softmax/exp/LDS-transpose of the current one runs
    ChunkFrags fa = loadChunk(0);
    for (int j0 = 0; j0 < 1024; j0 += 64){
        ChunkFrags fb = loadChunk(j0 + 32);
        compute(fa);
        if (j0 + 64 < 1024) fa = loadChunk(j0 + 64);
        compute(fb);
    }

    // normalize and emit attn_vec [B*Q, N*DH] bf16
    float rl[8];
#pragma unroll
    for (int v = 0; v < 8; ++v) rl[v] = 1.f / l[v];
#pragma unroll
    for (int c = 0; c < 4; ++c)
#pragma unroll
        for (int v = 0; v < 8; ++v){
            int row = i0 + v + 8 * hl;
            int col = n * 64 + c * 16 + l16;
            attnbf[((size_t)b * 1024 + row) * 1024 + col] = f2bf(O[c][v] * rl[v]);
        }
}

// ----------------------------- K3: output projection + residual + LayerNorm fused
__global__ void __launch_bounds__(256, 1) gemm_out_ln(
    const u16* __restrict__ attnbf, const u16* __restrict__ WoT,
    const float* __restrict__ win, const float* __restrict__ gamma,
    const float* __restrict__ beta, float* __restrict__ out)
{
    __shared__ float psum[16][8];
    __shared__ float psq[16][8];

    const int tid = threadIdx.x;
    const int wv  = tid >> 5, lane = tid & 31;
    const int l16 = lane & 15, hl = lane >> 4;
    const int r0  = blockIdx.x * 16;
    const int c0  = wv * 128;
    const int kb  = hl * 8;

    const u16* arow = attnbf + (size_t)(r0 + l16) * 1024;

    auto loadB8 = [&](v16u* B, int kk){
#pragma unroll
        for (int t = 0; t < 8; ++t){
            const u16* bp = WoT + (size_t)(c0 + t * 16 + l16) * 1024 + kk + hl * 16;
            B[t] = loadB32(bp);
        }
    };

    v8f acc[8];
#pragma unroll
    for (int t = 0; t < 8; ++t) acc[t] = vzero8();

    v16u A0 = loadA32(arow, 0, kb), A1;
    v16u B0[8], B1[8];
    loadB8(B0, 0);
    for (int kk = 0; kk < 1024; kk += 64){
        A1 = loadA32(arow, kk + 32, kb);
        loadB8(B1, kk + 32);
#pragma unroll
        for (int t = 0; t < 8; ++t) acc[t] = wmma_bf16(A0, B0[t], acc[t]);
        if (kk + 64 < 1024){
            A0 = loadA32(arow, kk + 64, kb);
            loadB8(B0, kk + 64);
        }
#pragma unroll
        for (int t = 0; t < 8; ++t) acc[t] = wmma_bf16(A1, B1[t], acc[t]);
    }

    // residual add (keep values in registers)
#pragma unroll
    for (int t = 0; t < 8; ++t)
#pragma unroll
        for (int v = 0; v < 8; ++v){
            int g = r0 + v + 8 * hl, col = c0 + t * 16 + l16;
            acc[t][v] += win[(size_t)g * 1024 + col];
        }

    // per-wave partial row sums -> LDS
#pragma unroll
    for (int v = 0; v < 8; ++v){
        float sm = 0.f, sq = 0.f;
#pragma unroll
        for (int t = 0; t < 8; ++t){ float x = acc[t][v]; sm += x; sq += x * x; }
        sm = rsum16(sm); sq = rsum16(sq);
        if (l16 == 0){ psum[v + 8 * hl][wv] = sm; psq[v + 8 * hl][wv] = sq; }
    }
    __syncthreads();

    float mu[8], rstd[8];
#pragma unroll
    for (int v = 0; v < 8; ++v){
        int rr = v + 8 * hl;
        float sm = 0.f, sq = 0.f;
#pragma unroll
        for (int wq = 0; wq < 8; ++wq){ sm += psum[rr][wq]; sq += psq[rr][wq]; }
        float m_ = sm * (1.f / 1024.f);
        float va = sq * (1.f / 1024.f) - m_ * m_;
        mu[v] = m_; rstd[v] = rsqrtf(va + 1e-5f);
    }

#pragma unroll
    for (int t = 0; t < 8; ++t)
#pragma unroll
        for (int v = 0; v < 8; ++v){
            int g = r0 + v + 8 * hl, col = c0 + t * 16 + l16;
            out[(size_t)g * 1024 + col] = (acc[t][v] - mu[v]) * rstd[v] * gamma[col] + beta[col];
        }
}

// ------------------------------------------------------------------ host launch
extern "C" void kernel_launch(void* const* d_in, const int* in_sizes, int n_in,
                              void* d_out, int out_size, void* d_ws, size_t ws_size,
                              hipStream_t stream)
{
    const float* w        = (const float*)d_in[0];
    const float* r_emb    = (const float*)d_in[1];
    const float* r_w_bias = (const float*)d_in[2];
    const float* r_bias   = (const float*)d_in[3];
    const float* Wq       = (const float*)d_in[4];
    const float* Wk       = (const float*)d_in[5];
    const float* Wv       = (const float*)d_in[6];
    const float* Wo       = (const float*)d_in[7];
    const float* gamma    = (const float*)d_in[8];
    const float* beta     = (const float*)d_in[9];
    float* out = (float*)d_out;

    char* ws = (char*)d_ws;
    size_t off = 0;
    auto alloc = [&](size_t bytes) -> void* {
        void* p = ws + off;
        off += (bytes + 255) & ~(size_t)255;
        return p;
    };
    u16* wbf    = (u16*)alloc((size_t)4096 * 1024 * 2);
    u16* WqT    = (u16*)alloc((size_t)1024 * 1024 * 2);
    u16* WkT    = (u16*)alloc((size_t)1024 * 1024 * 2);
    u16* WvT    = (u16*)alloc((size_t)1024 * 1024 * 2);
    u16* WoT    = (u16*)alloc((size_t)1024 * 1024 * 2);
    u16* rembbf = (u16*)alloc((size_t)1024 * 1024 * 2);
    u16* qbf    = (u16*)alloc((size_t)4096 * 1024 * 2);
    u16* rwbf   = (u16*)alloc((size_t)4096 * 1024 * 2);
    u16* kbf    = (u16*)alloc((size_t)4096 * 1024 * 2);
    u16* vTbf   = (u16*)alloc((size_t)4096 * 1024 * 2);
    u16* attnbf = (u16*)alloc((size_t)4096 * 1024 * 2);

    // K0: precision conversion + weight transposes
    k_cvt_bf16<<<(4096 * 1024) / 256, 256, 0, stream>>>(w, wbf, 4096 * 1024);
    k_cvt_bf16<<<(1024 * 1024) / 256, 256, 0, stream>>>(r_emb, rembbf, 1024 * 1024);
    dim3 tg(64, 64);
    k_transpose_cvt<<<tg, 256, 0, stream>>>(Wq, WqT, 1024, 1024);
    k_transpose_cvt<<<tg, 256, 0, stream>>>(Wk, WkT, 1024, 1024);
    k_transpose_cvt<<<tg, 256, 0, stream>>>(Wv, WvT, 1024, 1024);
    k_transpose_cvt<<<tg, 256, 0, stream>>>(Wo, WoT, 1024, 1024);

    // K1: QKV projections (WMMA, double-buffered)
    gemm_qkv<<<dim3(64, 12), 256, 0, stream>>>(wbf, WqT, WkT, WvT, r_w_bias,
                                               qbf, rwbf, kbf, vTbf);

    // K2: fused rel-attention + softmax + P@V (WMMA, flash-style, pipelined)
    attn_flash<<<512, 256, 0, stream>>>(qbf, rwbf, kbf, vTbf, rembbf, r_bias, attnbf);

    // K3: output projection + residual + LayerNorm (WMMA, double-buffered)
    gemm_out_ln<<<256, 256, 0, stream>>>(attnbf, WoT, w, gamma, beta, out);
}
